// LinearAttention_61366492725912
// MI455X (gfx1250) — compile-verified, matrix operand order checked by reading, not run on hardware
//
#include <hip/hip_runtime.h>
#include <hip/hip_bf16.h>

// ---------------------------------------------------------------------------
// Causal linear attention (ELU+1 features) as a HIERARCHICAL chunked scan.
// N=4, L=4096, H=12, D=M=64.
//
// Roofline: q,k,v,out are 48 MB each; the pure-streaming floor is ~8 us at
// 23.3 TB/s -- but a single-WG-per-(n,h) scan is only 48 workgroups, far too
// few to saturate HBM on MI455X.  So: split L into NBLK=8 blocks of 512 and
// run a 3-phase block scan with 384-way parallelism:
//   phase1: per (n,h,block) partial state  S_b = sum phi(k) v^T, ksum_b
//   phase2: exclusive scan of the 8 block records per (n,h)     (in-place)
//   phase3: per (n,h,block) causal chunk scan seeded with the scanned state
// All 64x64x64 chunk matmuls use v_wmma_f32_16x16x32_bf16 (f32 accumulate).
// Every WMMA operand is a row-major [lane][K] tile with K contiguous ->
// exactly two 16-B-aligned ds_load_b128 (LDS row stride 72 halfwords = 144 B).
// Transposed consumers (K^T as A, V/S as B) get transposed LDS copies written
// at producer time.  Falls back to a single-pass launch if d_ws is too small.
// ---------------------------------------------------------------------------

#define NBATCH 4
#define LSEQ   4096
#define NHEAD  12
#define NHTOT  (NBATCH * NHEAD)        /* 48 (n,h) pairs */
#define DDIM   64
#define CHUNK  64
#define NCHUNKS (LSEQ / CHUNK)         /* 64 */
#define NBLK   8                       /* L-blocks for the hierarchical scan */
#define CPB    (NCHUNKS / NBLK)        /* 8 chunks per block (512 tokens) */
#define RECSZ  (DDIM * DDIM + DDIM)    /* 4160 f32: [S 64x64][ksum 64] */
#define ROWSTR (NHEAD * DDIM)          /* 768 floats between consecutive l */
#define LDST   72                      /* LDS row stride: 144 B = 9*16 */
#define EPSV   1e-6f

typedef __bf16          v16bf __attribute__((ext_vector_type(16)));
typedef unsigned short  v8us  __attribute__((ext_vector_type(8)));  // 16 B
typedef float           v8f   __attribute__((ext_vector_type(8)));

static __device__ __forceinline__ unsigned short f2bf(float f) {
  unsigned u = __float_as_uint(f);
  u += 0x7fffu + ((u >> 16) & 1u);          // round-to-nearest-even
  return (unsigned short)(u >> 16);
}
static __device__ __forceinline__ float bf2f(unsigned short s) {
  return __uint_as_float(((unsigned)s) << 16);
}
static __device__ __forceinline__ float phi(float x) {   // elu(x)+1
  return x > 0.f ? x + 1.f : __expf(x);
}

// Row-major MxK operand gather.  ISA 16-bit A/B layout: lane = idx in M/N dim,
// lane>=16 takes the second K-half; element e <-> K = (e<8 ? e : e+8) + 8*half.
// Elements 0..7 / 8..15 are two contiguous 8xbf16 runs -> 2x ds_load_b128.
static __device__ __forceinline__ v16bf gather_mk(const unsigned short* p,
                                                  int m0, int k0, int lane) {
  const int m = lane & 15, hl = lane >> 4;
  const unsigned short* r = p + (m0 + m) * LDST + k0 + 8 * hl;
  union { v8us h[2]; v16bf b; } x;
  x.h[0] = *(const v8us*)(r);        // K = k0 + 8*hl + (0..7)
  x.h[1] = *(const v8us*)(r + 16);   // K = k0 + 8*hl + (16..23)
  return x.b;
}

#define WMMA_BF16(a, b, c) \
  __builtin_amdgcn_wmma_f32_16x16x32_bf16(false, (a), false, (b), (short)0, (c), false, false)

// ---------------------------------------------------------------------------
// Phase 1: per (n,h,block) partial state: S_b = sum_l phi(k_l) v_l^T (WMMA,
// f32 accumulators) and ksum_b[d] = sum_l phi(k_l)[d].
// ---------------------------------------------------------------------------
__global__ __launch_bounds__(256)
void lin_attn_phase1(const float* __restrict__ k, const float* __restrict__ v,
                     float* __restrict__ ws) {
  __shared__ __attribute__((aligned(16))) unsigned short Kt[DDIM * LDST]; // phi(k) [d][l]
  __shared__ __attribute__((aligned(16))) unsigned short Vt[DDIM * LDST]; // v      [m][l]

  const int tid  = threadIdx.x;
  const int lane = tid & 31;
  const int wv   = tid >> 5;
  const int col  = lane & 15, hl = lane >> 4;
  const int mj   = wv >> 1;              // db
  const int mn0  = (wv & 1) << 1;        // first ms
  const int nh   = blockIdx.x / NBLK;
  const int blk  = blockIdx.x % NBLK;
  const int nb   = nh / NHEAD, hh = nh % NHEAD;
  const size_t base = ((size_t)nb * LSEQ * NHEAD + (size_t)hh) * DDIM;
  const int l0base = blk * (CPB * CHUNK);

  v8f acc_s[2];
#pragma unroll
  for (int j = 0; j < 2; ++j)
#pragma unroll
    for (int e = 0; e < 8; ++e) acc_s[j][e] = 0.f;
  float kc = 0.f;                        // per-thread column sum (tid<64)

  for (int c = 0; c < CPB; ++c) {
    const int l0 = l0base + c * CHUNK;
    {
      const int row = tid >> 2;
      const int seg = (tid & 3) << 4;
      const size_t g = base + (size_t)(l0 + row) * ROWSTR + seg;
#pragma unroll
      for (int t4 = 0; t4 < 4; ++t4) {
        const int d0 = t4 * 4;
        const float4 fk = *(const float4*)(k + g + d0);
        const float4 fv = *(const float4*)(v + g + d0);
        Kt[(seg + d0 + 0) * LDST + row] = f2bf(phi(fk.x));
        Kt[(seg + d0 + 1) * LDST + row] = f2bf(phi(fk.y));
        Kt[(seg + d0 + 2) * LDST + row] = f2bf(phi(fk.z));
        Kt[(seg + d0 + 3) * LDST + row] = f2bf(phi(fk.w));
        Vt[(seg + d0 + 0) * LDST + row] = f2bf(fv.x);
        Vt[(seg + d0 + 1) * LDST + row] = f2bf(fv.y);
        Vt[(seg + d0 + 2) * LDST + row] = f2bf(fv.z);
        Vt[(seg + d0 + 3) * LDST + row] = f2bf(fv.w);
      }
      if (c + 1 < CPB) {
        __builtin_prefetch(k + g + (size_t)CHUNK * ROWSTR, 0, 1);
        __builtin_prefetch(v + g + (size_t)CHUNK * ROWSTR, 0, 1);
      }
    }
    __syncthreads();

    if (tid < DDIM)
      for (int i = 0; i < CHUNK; ++i) kc += bf2f(Kt[tid * LDST + i]);

    const v16bf ak0 = gather_mk(Kt, mj << 4, 0,  lane);
    const v16bf ak1 = gather_mk(Kt, mj << 4, 32, lane);
#pragma unroll
    for (int j = 0; j < 2; ++j) {
      const int ms = mn0 + j;
      acc_s[j] = WMMA_BF16(ak0, gather_mk(Vt, ms << 4, 0,  lane), acc_s[j]);
      acc_s[j] = WMMA_BF16(ak1, gather_mk(Vt, ms << 4, 32, lane), acc_s[j]);
    }
    __syncthreads();   // before next chunk overwrites Kt/Vt
  }

  float* rec = ws + (size_t)(nh * NBLK + blk) * RECSZ;
#pragma unroll
  for (int j = 0; j < 2; ++j) {
    const int ms = mn0 + j;
#pragma unroll
    for (int r = 0; r < 8; ++r)
      rec[((mj << 4) + r + 8 * hl) * DDIM + (ms << 4) + col] = acc_s[j][r];
  }
  if (tid < DDIM) rec[DDIM * DDIM + tid] = kc;
}

// ---------------------------------------------------------------------------
// Phase 2: in-place exclusive scan over the NBLK block records per (n,h).
// ---------------------------------------------------------------------------
__global__ __launch_bounds__(256)
void lin_attn_phase2(float* __restrict__ ws) {
  const int nh  = blockIdx.x;
  const int tid = threadIdx.x;
  for (int e = tid; e < RECSZ; e += 256) {
    float acc = 0.f;
    for (int b = 0; b < NBLK; ++b) {
      float* p = ws + (size_t)(nh * NBLK + b) * RECSZ + e;
      const float t = *p;
      *p = acc;                 // exclusive prefix: state at block entry
      acc += t;
    }
  }
}

// ---------------------------------------------------------------------------
// Phase 3: causal chunk scan over one L-block, seeded from the scanned state.
// Also serves as the single-pass fallback (states==nullptr, nblk=1).
// ---------------------------------------------------------------------------
__global__ __launch_bounds__(256)
void lin_attn_phase3(const float* __restrict__ q,
                     const float* __restrict__ k,
                     const float* __restrict__ v,
                     float* __restrict__ out,
                     const float* __restrict__ states,
                     int nblk, int cpb) {
  __shared__ __attribute__((aligned(16))) unsigned short Qs[CHUNK * LDST]; // phi(q) [l][d]
  __shared__ __attribute__((aligned(16))) unsigned short Ks[CHUNK * LDST]; // phi(k) [l][d]
  __shared__ __attribute__((aligned(16))) unsigned short Kt[DDIM  * LDST]; // phi(k) [d][l]
  __shared__ __attribute__((aligned(16))) unsigned short Vt[DDIM  * LDST]; // v      [m][l]
  __shared__ __attribute__((aligned(16))) unsigned short Ps[CHUNK * LDST]; // masked P [l][l']
  __shared__ __attribute__((aligned(16))) unsigned short St[DDIM  * LDST]; // S_prev [m][d]
  __shared__ float zdot[CHUNK];   // qf_i . ksum_prev
  __shared__ float zinv[CHUNK];   // 1/z_i
  __shared__ float ksum[DDIM];    // running column-sum of phi(k)

  const int tid  = threadIdx.x;
  const int lane = tid & 31;
  const int wv   = tid >> 5;
  const int col  = lane & 15, hl = lane >> 4;
  const int mj   = wv >> 1;              // rb (P,O) and db (S)
  const int mn0  = (wv & 1) << 1;        // first minor index (cb / ms)
  const int nh   = blockIdx.x / nblk;
  const int blk  = blockIdx.x % nblk;
  const int nb   = nh / NHEAD, hh = nh % NHEAD;
  const size_t base = ((size_t)nb * LSEQ * NHEAD + (size_t)hh) * DDIM;

  // Seed state from the scanned block-entry record (or zero for fallback).
  v8f acc_s[2];
  if (states) {
    const float* rec = states + (size_t)(nh * nblk + blk) * RECSZ;
#pragma unroll
    for (int j = 0; j < 2; ++j) {
      const int ms = mn0 + j;
#pragma unroll
      for (int r = 0; r < 8; ++r)
        acc_s[j][r] = rec[((mj << 4) + r + 8 * hl) * DDIM + (ms << 4) + col];
    }
    if (tid < DDIM) ksum[tid] = rec[DDIM * DDIM + tid];
  } else {
#pragma unroll
    for (int j = 0; j < 2; ++j)
#pragma unroll
      for (int e = 0; e < 8; ++e) acc_s[j][e] = 0.f;
    if (tid < DDIM) ksum[tid] = 0.f;
  }
  __syncthreads();

  for (int c = 0; c < cpb; ++c) {
    const int l0 = (blk * cpb + c) * CHUNK;

    if (tid < CHUNK) zdot[tid] = 0.f;
    __syncthreads();

    // ---- load + featurize: thread -> (row = tid/4, 16-wide d-segment) ----
    {
      const int row = tid >> 2;
      const int seg = (tid & 3) << 4;
      const size_t g = base + (size_t)(l0 + row) * ROWSTR + seg;
      unsigned short qb[16], kb[16], vb[16];
      float part = 0.f;
#pragma unroll
      for (int t4 = 0; t4 < 4; ++t4) {
        const int d0 = t4 * 4;
        const float4 fq = *(const float4*)(q + g + d0);
        const float4 fk = *(const float4*)(k + g + d0);
        const float4 fv = *(const float4*)(v + g + d0);
        const float q0 = phi(fq.x), q1 = phi(fq.y), q2 = phi(fq.z), q3 = phi(fq.w);
        const float ka = phi(fk.x), kc = phi(fk.y), kd = phi(fk.z), ke = phi(fk.w);
        qb[d0 + 0] = f2bf(q0); qb[d0 + 1] = f2bf(q1);
        qb[d0 + 2] = f2bf(q2); qb[d0 + 3] = f2bf(q3);
        kb[d0 + 0] = f2bf(ka); kb[d0 + 1] = f2bf(kc);
        kb[d0 + 2] = f2bf(kd); kb[d0 + 3] = f2bf(ke);
        vb[d0 + 0] = f2bf(fv.x); vb[d0 + 1] = f2bf(fv.y);
        vb[d0 + 2] = f2bf(fv.z); vb[d0 + 3] = f2bf(fv.w);
        part += q0 * ksum[seg + d0 + 0] + q1 * ksum[seg + d0 + 1]
              + q2 * ksum[seg + d0 + 2] + q3 * ksum[seg + d0 + 3];
      }
#pragma unroll
      for (int j = 0; j < 16; ++j) {             // row-major (merged stores)
        Qs[row * LDST + seg + j] = qb[j];
        Ks[row * LDST + seg + j] = kb[j];
      }
#pragma unroll
      for (int j = 0; j < 16; ++j) {             // transposed copies
        Kt[(seg + j) * LDST + row] = kb[j];
        Vt[(seg + j) * LDST + row] = vb[j];
      }
      atomicAdd(&zdot[row], part);               // ds_add_f32
      if (c + 1 < cpb) {                         // global_prefetch next chunk
        __builtin_prefetch(q + g + (size_t)CHUNK * ROWSTR, 0, 1);
        __builtin_prefetch(k + g + (size_t)CHUNK * ROWSTR, 0, 1);
        __builtin_prefetch(v + g + (size_t)CHUNK * ROWSTR, 0, 1);
      }
    }
    __syncthreads();

    // ---- running k-feature column-sum (for next chunk's zdot) ----
    if (tid < DDIM) {
      float s = ksum[tid];
      for (int i = 0; i < CHUNK; ++i) s += bf2f(Kt[tid * LDST + i]);
      ksum[tid] = s;
    }

    // A-operands shared by both of this wave's tiles (and across P/O phases).
    const v16bf aq0 = gather_mk(Qs, mj << 4, 0,  lane);
    const v16bf aq1 = gather_mk(Qs, mj << 4, 32, lane);

    // ---- P = Qf Kf^T, causal-masked, stored bf16 row-major ----
#pragma unroll
    for (int j = 0; j < 2; ++j) {
      const int rb = mj, cb = mn0 + j;
      const int r0 = rb << 4, c0 = cb << 4;
      if (cb > rb) {                 // strictly upper tile: zero
#pragma unroll
        for (int r = 0; r < 8; ++r)
          Ps[(r0 + r + 8 * hl) * LDST + c0 + col] = 0;
      } else {
        v8f pc = {};
        pc = WMMA_BF16(aq0, gather_mk(Ks, c0, 0,  lane), pc);
        pc = WMMA_BF16(aq1, gather_mk(Ks, c0, 32, lane), pc);
#pragma unroll
        for (int r = 0; r < 8; ++r) {
          float pv = pc[r];
          if (cb == rb && col > r + 8 * hl) pv = 0.f;   // mask j>i on diagonal
          Ps[(r0 + r + 8 * hl) * LDST + c0 + col] = f2bf(pv);
        }
      }
    }

    // ---- snapshot S_prev (pre-update) transposed: St[m][d] ----
#pragma unroll
    for (int j = 0; j < 2; ++j) {
      const int db = mj, ms = mn0 + j;
#pragma unroll
      for (int r = 0; r < 8; ++r)
        St[((ms << 4) + col) * LDST + (db << 4) + r + 8 * hl] = f2bf(acc_s[j][r]);
    }
    __syncthreads();

    // ---- normalizer: z_i = zdot_i + masked rowsum of P ----
    if (tid < CHUNK) {
      float rs = zdot[tid];
      for (int jj = 0; jj <= tid; ++jj) rs += bf2f(Ps[tid * LDST + jj]);
      zinv[tid] = 1.f / (rs + EPSV);
    }

    // ---- O = Qf @ S_prev + Pm @ V  and  S += Kf^T @ V ----
    const v16bf ap0 = gather_mk(Ps, mj << 4, 0,  lane);
    const v16bf ap1 = gather_mk(Ps, mj << 4, 32, lane);
    const v16bf ak0 = gather_mk(Kt, mj << 4, 0,  lane);
    const v16bf ak1 = gather_mk(Kt, mj << 4, 32, lane);
    v8f oc[2];
#pragma unroll
    for (int j = 0; j < 2; ++j) {
      const int ms = mn0 + j;
      const v16bf bv0 = gather_mk(Vt, ms << 4, 0,  lane);
      const v16bf bv1 = gather_mk(Vt, ms << 4, 32, lane);
      v8f cc = {};
      cc = WMMA_BF16(aq0, gather_mk(St, ms << 4, 0,  lane), cc);
      cc = WMMA_BF16(aq1, gather_mk(St, ms << 4, 32, lane), cc);
      cc = WMMA_BF16(ap0, bv0, cc);
      cc = WMMA_BF16(ap1, bv1, cc);
      oc[j] = cc;
      acc_s[j] = WMMA_BF16(ak0, bv0, acc_s[j]);
      acc_s[j] = WMMA_BF16(ak1, bv1, acc_s[j]);
    }
    __syncthreads();   // zinv ready; also closes this chunk's LDS reads

    // ---- write O / z ----
#pragma unroll
    for (int j = 0; j < 2; ++j) {
      const int rb = mj, ms = mn0 + j;
#pragma unroll
      for (int r = 0; r < 8; ++r) {
        const int i = (rb << 4) + r + 8 * hl;
        out[base + (size_t)(l0 + i) * ROWSTR + (ms << 4) + col] = oc[j][r] * zinv[i];
      }
    }
  }
}

extern "C" void kernel_launch(void* const* d_in, const int* in_sizes, int n_in,
                              void* d_out, int out_size, void* d_ws, size_t ws_size,
                              hipStream_t stream) {
  (void)in_sizes; (void)n_in; (void)out_size;
  const float* q = (const float*)d_in[0];
  const float* k = (const float*)d_in[1];
  const float* v = (const float*)d_in[2];
  float* out = (float*)d_out;

  const size_t need = (size_t)NHTOT * NBLK * RECSZ * sizeof(float);  // ~6.4 MB
  if (d_ws != nullptr && ws_size >= need) {
    float* ws = (float*)d_ws;
    lin_attn_phase1<<<NHTOT * NBLK, 256, 0, stream>>>(k, v, ws);
    lin_attn_phase2<<<NHTOT, 256, 0, stream>>>(ws);
    lin_attn_phase3<<<NHTOT * NBLK, 256, 0, stream>>>(q, k, v, out, ws, NBLK, CPB);
  } else {
    // Workspace too small: single-pass fallback (48 WGs, zero initial state).
    lin_attn_phase3<<<NHTOT, 256, 0, stream>>>(q, k, v, out, nullptr, 1, NCHUNKS);
  }
}
